// PerViewAttentionPool3d_4569845203563
// MI455X (gfx1250) — matrix operand order repaired and checked by reading, not compiled, not run on hardware
//
#include <hip/hip_runtime.h>
#include <hip/hip_bf16.h>
#include <stdint.h>

typedef _Float16 v16h __attribute__((ext_vector_type(16)));
typedef _Float16 v8h  __attribute__((ext_vector_type(8)));
typedef _Float16 v4h  __attribute__((ext_vector_type(4)));
typedef float    v8f  __attribute__((ext_vector_type(8)));

#define D       512
#define KOFF    27
#define HEADS   8
#define DH      64
#define VIEWS   6
#define BSZ     4
#define ROWS    64          // rows per block tile
#define LDA     520         // LDS A stride in halves (1040B row -> conflict-free b128 ds loads)
#define BSTR    40          // LDS B stride in halves per n (80B -> lanes hit disjoint 4-bank groups)

__device__ inline v8f zero8() {
  v8f z;
#pragma unroll
  for (int j = 0; j < 8; ++j) z[j] = 0.0f;
  return z;
}

// ---------------- K0a: W_cpe [27][c][n] f32 -> tiled f16 [27][kc][n][32] where k = kc*32+c5
__global__ void k_conv_wcpe(const float* __restrict__ w, _Float16* __restrict__ o) {
  int i = blockIdx.x * blockDim.x + threadIdx.x;     // over 27*512*512
  if (i >= KOFF * D * D) return;
  int c5  = i & 31;
  int n   = (i >> 5) & 511;
  int kc  = (i >> 14) & 15;
  int off = i >> 18;
  // B[k][n] = W_cpe[off][k][n]
  o[i] = (_Float16)w[(off << 18) + ((kc * 32 + c5) << 9) + n];
}

// ---------------- K0b: W [n][k] f32 (used as x @ W^T) -> tiled f16 [kc][n][32]
__global__ void k_conv_tiled(const float* __restrict__ w, _Float16* __restrict__ o) {
  int i = blockIdx.x * blockDim.x + threadIdx.x;     // over 512*512
  if (i >= D * D) return;
  int c5 = i & 31;
  int n  = (i >> 5) & 511;
  int kc = i >> 14;
  // B[k][n] = W[n][k]
  o[i] = (_Float16)w[(n << 9) + (kc << 5) + c5];
}

// ---------------- Shared GEMM body:
// Block: 256 threads (8 waves), tile = 64 rows x 512 cols; 2 row-waves x 4 col-waves;
// wave = 32 rows (2 A-tiles) x 128 cols (8 B-tiles) = 16 accum tiles.
// B double-buffered in LDS: next slice's 8 global_load_b128 are issued before the 16 WMMAs,
// stored after them -> global latency hidden behind compute, one barrier per k-step.
__device__ __forceinline__ void gemm512(const _Float16* __restrict__ Wt,
                                        _Float16* Bsh0, _Float16* Bsh1,
                                        const _Float16* Ash, v8f* acc,
                                        int tid, int a0row, int a1row,
                                        int kAddA, int bOff0) {
  // prologue: stage kt=0 into Bsh0 (load-all then store-all: one latency)
  {
    v8h t[8];
#pragma unroll
    for (int j = 0; j < 8; ++j)
      t[j] = *(const v8h*)(Wt + (size_t)(j * 256 + tid) * 8);
#pragma unroll
    for (int j = 0; j < 8; ++j) {
      int u = j * 256 + tid;
      int n = u >> 2, q = u & 3;
      *(v8h*)&Bsh0[n * BSTR + q * 8] = t[j];
    }
  }
  __syncthreads();
  for (int kt = 0; kt < 16; ++kt) {
    _Float16* bufc = (kt & 1) ? Bsh1 : Bsh0;
    _Float16* bufn = (kt & 1) ? Bsh0 : Bsh1;
    v8h t[8];
    if (kt < 15) {
      const _Float16* src = Wt + ((size_t)(kt + 1) << 14);
#pragma unroll
      for (int j = 0; j < 8; ++j)
        t[j] = *(const v8h*)(src + (size_t)(j * 256 + tid) * 8);
    }
    union { v16h v; v8h p[2]; } af0, af1, bcur, bnxt;
    af0.p[0] = *(const v8h*)&Ash[a0row * LDA + kt * 32 + kAddA];
    af0.p[1] = *(const v8h*)&Ash[a0row * LDA + kt * 32 + kAddA + 16];
    af1.p[0] = *(const v8h*)&Ash[a1row * LDA + kt * 32 + kAddA];
    af1.p[1] = *(const v8h*)&Ash[a1row * LDA + kt * 32 + kAddA + 16];
    bcur.p[0] = *(const v8h*)&bufc[bOff0];
    bcur.p[1] = *(const v8h*)&bufc[bOff0 + 16];
#pragma unroll
    for (int nt = 0; nt < 8; ++nt) {
      if (nt < 7) {
        const _Float16* bp = &bufc[bOff0 + (nt + 1) * (16 * BSTR)];
        bnxt.p[0] = *(const v8h*)(bp);
        bnxt.p[1] = *(const v8h*)(bp + 16);
      }
      acc[nt] = __builtin_amdgcn_wmma_f32_16x16x32_f16(
          false, af0.v, false, bcur.v, (short)0, acc[nt], false, false);
      acc[nt + 8] = __builtin_amdgcn_wmma_f32_16x16x32_f16(
          false, af1.v, false, bcur.v, (short)0, acc[nt + 8], false, false);
      bcur = bnxt;
    }
    if (kt < 15) {
#pragma unroll
      for (int j = 0; j < 8; ++j) {
        int u = j * 256 + tid;
        int n = u >> 2, q = u & 3;
        *(v8h*)&bufn[n * BSTR + q * 8] = t[j];
      }
    }
    __syncthreads();
  }
}

// ---------------- K1: fused cpe (27 gathered WMMA GEMMs) + W_lin GEMM + LN + residual -> x_h
__global__ __launch_bounds__(256)
void k_cpe_ln(const float* __restrict__ feats,
              const int* __restrict__ nbr_idx,
              const unsigned char* __restrict__ nbr_valid,
              const _Float16* __restrict__ Wcpe,   // tiled [27][16][512][32] f16
              const float* __restrict__ b_cpe,
              const _Float16* __restrict__ Wlin,   // tiled [16][512][32] f16
              const float* __restrict__ b_lin,
              const float* __restrict__ ln_g,
              const float* __restrict__ ln_b,
              _Float16* __restrict__ xh,
              int T) {
  __shared__ _Float16 Ash[ROWS * LDA];        // 66,560 B
  __shared__ _Float16 Bsh[2][D * BSTR];       // 2 x 40,960 B
  __shared__ int   s_idx[ROWS];
  __shared__ float s_red[256], s_red2[256];
  __shared__ float s_mu[ROWS], s_rs[ROWS];

  const int tid    = threadIdx.x;
  const int lane   = tid & 31;
  const int wave   = tid >> 5;
  const int wrow   = wave & 1;     // 0..1 (32 rows each)
  const int wcol   = wave >> 1;    // 0..3 (128 cols each)
  const int l16    = lane & 15;
  const int hiHalf = lane >> 4;
  const int row0   = blockIdx.x * ROWS;

  const int a0row = wrow * 32 + l16;
  const int a1row = a0row + 16;
  const int kAddA = hiHalf ? 8 : 0;
  const int colBase = wcol * 128 + l16;
  const int bOff0   = colBase * BSTR + (hiHalf ? 8 : 0);

  v8f acc[16];                     // [nt] = rows 0..15, [nt+8] = rows 16..31
#pragma unroll
  for (int nt = 0; nt < 16; ++nt) acc[nt] = zero8();

  for (int off = 0; off < KOFF; ++off) {
    if (tid < ROWS) {
      int rg = row0 + tid;
      int idx = -1;
      if (rg < T && nbr_valid[rg * KOFF + off]) idx = nbr_idx[rg * KOFF + off];
      s_idx[tid] = idx;
    }
    __syncthreads();
    // stage gathered rows (f32 -> f16) into LDS A; chunks of 8 float4 (clause loads)
#pragma unroll
    for (int chunk = 0; chunk < 4; ++chunk) {
      float4 t[8];
#pragma unroll
      for (int j = 0; j < 8; ++j) {
        int i = chunk * 2048 + j * 256 + tid;
        int r = i >> 7, c4 = i & 127;
        int idx = s_idx[r];
        t[j] = make_float4(0.f, 0.f, 0.f, 0.f);
        if (idx >= 0) t[j] = ((const float4*)(feats + (size_t)idx * D))[c4];
      }
#pragma unroll
      for (int j = 0; j < 8; ++j) {
        int i = chunk * 2048 + j * 256 + tid;
        int r = i >> 7, c4 = i & 127;
        v4h h; h[0] = (_Float16)t[j].x; h[1] = (_Float16)t[j].y;
               h[2] = (_Float16)t[j].z; h[3] = (_Float16)t[j].w;
        *(v4h*)&Ash[r * LDA + c4 * 4] = h;
      }
    }
    __syncthreads();

    gemm512(Wcpe + ((size_t)off << 18), Bsh[0], Bsh[1], Ash, acc,
            tid, a0row, a1row, kAddA, bOff0);
  }

  // write cpe result (+b_cpe) back to LDS A (f16) for the W_lin GEMM
  {
    const int mrow = wrow * 32 + hiHalf * 8;
#pragma unroll
    for (int nt = 0; nt < 8; ++nt) {
      int col = colBase + nt * 16;
      float bias = b_cpe[col];
#pragma unroll
      for (int r = 0; r < 8; ++r) {
        Ash[(mrow + r) * LDA + col]      = (_Float16)(acc[nt][r] + bias);
        Ash[(mrow + 16 + r) * LDA + col] = (_Float16)(acc[nt + 8][r] + bias);
      }
    }
  }
  __syncthreads();

  v8f acc2[16];
#pragma unroll
  for (int nt = 0; nt < 16; ++nt) acc2[nt] = zero8();
  gemm512(Wlin, Bsh[0], Bsh[1], Ash, acc2, tid, a0row, a1row, kAddA, bOff0);

  // unpack (+b_lin) into LDS A for layernorm
  {
    const int mrow = wrow * 32 + hiHalf * 8;
#pragma unroll
    for (int nt = 0; nt < 8; ++nt) {
      int col = colBase + nt * 16;
      float bias = b_lin[col];
#pragma unroll
      for (int r = 0; r < 8; ++r) {
        Ash[(mrow + r) * LDA + col]      = (_Float16)(acc2[nt][r] + bias);
        Ash[(mrow + 16 + r) * LDA + col] = (_Float16)(acc2[nt + 8][r] + bias);
      }
    }
  }
  __syncthreads();
  // layernorm stats: 4 threads per row (128 cols each)
  {
    int r = tid >> 2, seg = tid & 3;
    float s = 0.f, q = 0.f;
    const _Float16* rp = &Ash[r * LDA + seg * 128];
    for (int c = 0; c < 128; ++c) { float v = (float)rp[c]; s += v; q += v * v; }
    s_red[tid] = s; s_red2[tid] = q;
  }
  __syncthreads();
  if (tid < ROWS) {
    float s = 0.f, q = 0.f;
    for (int j = 0; j < 4; ++j) { s += s_red[tid * 4 + j]; q += s_red2[tid * 4 + j]; }
    float mu  = s / (float)D;
    float var = q / (float)D - mu * mu;
    s_mu[tid] = mu;
    s_rs[tid] = rsqrtf(var + 1e-5f);
  }
  __syncthreads();
  // residual add + store x as f16 (chunked: clause the feats loads)
#pragma unroll
  for (int chunk = 0; chunk < 4; ++chunk) {
    float4 t[8];
#pragma unroll
    for (int j = 0; j < 8; ++j) {
      int i = chunk * 2048 + j * 256 + tid;
      int r = i >> 7, c4 = i & 127;
      int rg = row0 + r;
      t[j] = make_float4(0.f, 0.f, 0.f, 0.f);
      if (rg < T) t[j] = ((const float4*)(feats + (size_t)rg * D))[c4];
    }
#pragma unroll
    for (int j = 0; j < 8; ++j) {
      int i = chunk * 2048 + j * 256 + tid;
      int r = i >> 7, c4 = i & 127;
      int rg = row0 + r;
      if (rg >= T) continue;
      float mu = s_mu[r], rs = s_rs[r];
      int c = c4 * 4;
      v4h h;
      h[0] = (_Float16)(((float)Ash[r * LDA + c + 0] - mu) * rs * ln_g[c + 0] + ln_b[c + 0] + t[j].x);
      h[1] = (_Float16)(((float)Ash[r * LDA + c + 1] - mu) * rs * ln_g[c + 1] + ln_b[c + 1] + t[j].y);
      h[2] = (_Float16)(((float)Ash[r * LDA + c + 2] - mu) * rs * ln_g[c + 2] + ln_b[c + 2] + t[j].z);
      h[3] = (_Float16)(((float)Ash[r * LDA + c + 3] - mu) * rs * ln_g[c + 3] + ln_b[c + 3] + t[j].w);
      *(v4h*)&xh[(size_t)rg * D + c] = h;
    }
  }
}

// ---------------- K3: K/V projections, same tiling; one staged A tile, two GEMM passes
__global__ __launch_bounds__(256)
void k_kv(const _Float16* __restrict__ xh,
          const _Float16* __restrict__ Wk, const float* __restrict__ bk,
          const _Float16* __restrict__ Wv, const float* __restrict__ bv,
          _Float16* __restrict__ kk, _Float16* __restrict__ vv, int T) {
  __shared__ _Float16 Ash[ROWS * LDA];
  __shared__ _Float16 Bsh[2][D * BSTR];
  const int tid    = threadIdx.x;
  const int lane   = tid & 31;
  const int wave   = tid >> 5;
  const int wrow   = wave & 1;
  const int wcol   = wave >> 1;
  const int l16    = lane & 15;
  const int hiHalf = lane >> 4;
  const int row0   = blockIdx.x * ROWS;

  const int a0row = wrow * 32 + l16;
  const int a1row = a0row + 16;
  const int kAddA = hiHalf ? 8 : 0;
  const int colBase = wcol * 128 + l16;
  const int bOff0   = colBase * BSTR + (hiHalf ? 8 : 0);
  const int mrow    = wrow * 32 + hiHalf * 8;

  // stage 64 rows of x_h into LDS A (chunks of 8 v8h: clause loads)
#pragma unroll
  for (int chunk = 0; chunk < 2; ++chunk) {
    v8h t[8];
#pragma unroll
    for (int j = 0; j < 8; ++j) {
      int i = chunk * 2048 + j * 256 + tid;
      int r = i >> 6, c8 = i & 63;
      int rg = row0 + r;
#pragma unroll
      for (int x = 0; x < 8; ++x) t[j][x] = (_Float16)0;
      if (rg < T) t[j] = *(const v8h*)(xh + (size_t)rg * D + c8 * 8);
    }
#pragma unroll
    for (int j = 0; j < 8; ++j) {
      int i = chunk * 2048 + j * 256 + tid;
      int r = i >> 6, c8 = i & 63;
      *(v8h*)&Ash[r * LDA + c8 * 8] = t[j];
    }
  }
  __syncthreads();

  for (int pass = 0; pass < 2; ++pass) {
    const _Float16* W = pass ? Wv : Wk;
    const float*    b = pass ? bv : bk;
    _Float16*       O = pass ? vv : kk;
    v8f acc[16];
#pragma unroll
    for (int nt = 0; nt < 16; ++nt) acc[nt] = zero8();
    gemm512(W, Bsh[0], Bsh[1], Ash, acc, tid, a0row, a1row, kAddA, bOff0);
#pragma unroll
    for (int nt = 0; nt < 8; ++nt) {
      int col = colBase + nt * 16;
      float bias = b[col];
#pragma unroll
      for (int r = 0; r < 8; ++r) {
        int rg0 = row0 + mrow + r;
        int rg1 = rg0 + 16;
        if (rg0 < T) O[(size_t)rg0 * D + col] = (_Float16)(acc[nt][r] + bias);
        if (rg1 < T) O[(size_t)rg1 * D + col] = (_Float16)(acc[nt + 8][r] + bias);
      }
    }
  }
}

// ---------------- K2: per-(b,v) visible mean pooling -> q_in [24][512] f32
__global__ __launch_bounds__(256)
void k_pool(const _Float16* __restrict__ xh, const unsigned char* __restrict__ vis,
            float* __restrict__ qin, int npts) {
  int b = blockIdx.x / VIEWS, v = blockIdx.x % VIEWS;
  int tid = threadIdx.x;
  __shared__ float sc[256];
  const unsigned char* vp = vis + (size_t)b * npts * VIEWS + v;
  int cnt = 0;
  for (int n = tid; n < npts; n += 256) cnt += vp[(size_t)n * VIEWS];
  sc[tid] = (float)cnt;
  __syncthreads();
  for (int s = 128; s > 0; s >>= 1) { if (tid < s) sc[tid] += sc[tid + s]; __syncthreads(); }
  float count = sc[0];
  __syncthreads();
  float a0 = 0.f, a1 = 0.f;
  const _Float16* xb = xh + (size_t)b * npts * D;
  for (int n = 0; n < npts; ++n) {
    if (vp[(size_t)n * VIEWS]) {
      a0 += (float)xb[(size_t)n * D + tid];
      a1 += (float)xb[(size_t)n * D + tid + 256];
    }
  }
  float inv = 1.0f / count;
  qin[(size_t)blockIdx.x * D + tid]       = a0 * inv;
  qin[(size_t)blockIdx.x * D + tid + 256] = a1 * inv;
}

// ---------------- K4: q = (q_in @ Wq^T + bq) * dh^-0.5   (tiny, scalar f32)
__global__ void k_qproj(const float* __restrict__ qin, const float* __restrict__ Wq,
                        const float* __restrict__ bq, float* __restrict__ q) {
  int bv = blockIdx.x, tid = threadIdx.x;
  const float* x = qin + (size_t)bv * D;
  for (int d = tid; d < D; d += 256) {
    const float* w = Wq + (size_t)d * D;
    float s = 0.f;
    for (int c = 0; c < D; ++c) s += x[c] * w[c];
    q[(size_t)bv * D + d] = (s + bq[d]) * 0.125f;   // 64^-0.5
  }
}

// ---------------- K5: attention per (b,h,v): masked softmax over npts keys
__global__ __launch_bounds__(256)
void k_attn(const float* __restrict__ q, const _Float16* __restrict__ kk,
            const _Float16* __restrict__ vv, const unsigned char* __restrict__ vis,
            float* __restrict__ sg_all, float* __restrict__ oatt, int npts) {
  int b = blockIdx.x / (HEADS * VIEWS);
  int rem = blockIdx.x % (HEADS * VIEWS);
  int h = rem / VIEWS;
  int v = rem % VIEWS;
  int tid = threadIdx.x;
  __shared__ float qsh[DH];
  __shared__ float red[256];
  float* sg = sg_all + (size_t)blockIdx.x * npts;
  if (tid < DH) qsh[tid] = q[(size_t)(b * VIEWS + v) * D + h * DH + tid];
  __syncthreads();

  const unsigned char* vp = vis + (size_t)b * npts * VIEWS + v;
  const _Float16* kbase = kk + (size_t)b * npts * D + h * DH;
  float m = -3.0e38f;
  for (int n = tid; n < npts; n += 256) {
    float s = -3.0e38f;
    if (vp[(size_t)n * VIEWS]) {
      const _Float16* kp = kbase + (size_t)n * D;
      float acc = 0.f;
#pragma unroll
      for (int d0 = 0; d0 < DH; d0 += 8) {
        v8h kv8 = *(const v8h*)(kp + d0);
#pragma unroll
        for (int j = 0; j < 8; ++j) acc += (float)kv8[j] * qsh[d0 + j];
      }
      s = acc;
    }
    sg[n] = s;
    m = fmaxf(m, s);
  }
  red[tid] = m; __syncthreads();
  for (int s2 = 128; s2 > 0; s2 >>= 1) { if (tid < s2) red[tid] = fmaxf(red[tid], red[tid + s2]); __syncthreads(); }
  float M = red[0];
  __syncthreads();
  __threadfence();
  float l = 0.f;
  for (int n = tid; n < npts; n += 256) {
    float e = __expf(sg[n] - M);
    sg[n] = e;
    l += e;
  }
  red[tid] = l; __syncthreads();
  for (int s2 = 128; s2 > 0; s2 >>= 1) { if (tid < s2) red[tid] += red[tid + s2]; __syncthreads(); }
  float L = red[0];
  __syncthreads();
  __threadfence();
  int d = tid & 63, j = tid >> 6;
  const _Float16* vbase = vv + (size_t)b * npts * D + h * DH;
  float a = 0.f;
  for (int n = j; n < npts; n += 4)
    a += sg[n] * (float)vbase[(size_t)n * D + d];
  red[tid] = a; __syncthreads();
  if (j == 0) {
    float t = red[d] + red[d + 64] + red[d + 128] + red[d + 192];
    oatt[(size_t)(b * VIEWS + v) * D + h * DH + d] = t / L;
  }
}

// ---------------- K6: output projection (tiny, scalar f32)
__global__ void k_cproj(const float* __restrict__ oatt, const float* __restrict__ Wc,
                        const float* __restrict__ bc, float* __restrict__ out) {
  int bv = blockIdx.x, tid = threadIdx.x;
  const float* x = oatt + (size_t)bv * D;
  for (int o = tid; o < D; o += 256) {
    const float* w = Wc + (size_t)o * D;
    float s = 0.f;
    for (int c = 0; c < D; ++c) s += x[c] * w[c];
    out[(size_t)bv * D + o] = s + bc[o];
  }
}

extern "C" void kernel_launch(void* const* d_in, const int* in_sizes, int n_in,
                              void* d_out, int out_size, void* d_ws, size_t ws_size,
                              hipStream_t stream) {
  const float* feats = (const float*)d_in[0];
  const float* Wcpe  = (const float*)d_in[1];
  const float* b_cpe = (const float*)d_in[2];
  const float* Wlin  = (const float*)d_in[3];
  const float* b_lin = (const float*)d_in[4];
  const float* ln_g  = (const float*)d_in[5];
  const float* ln_b  = (const float*)d_in[6];
  const float* Wq    = (const float*)d_in[7];
  const float* bq    = (const float*)d_in[8];
  const float* Wk    = (const float*)d_in[9];
  const float* bk    = (const float*)d_in[10];
  const float* Wv    = (const float*)d_in[11];
  const float* bv    = (const float*)d_in[12];
  const float* Wc    = (const float*)d_in[13];
  const float* bc    = (const float*)d_in[14];
  const int*   nbr_idx = (const int*)d_in[15];
  const unsigned char* nbr_valid = (const unsigned char*)d_in[16];
  const unsigned char* vis       = (const unsigned char*)d_in[17];

  const int T    = in_sizes[0] / D;       // 100000
  const int npts = T / BSZ;               // 25000

  // workspace carve-out (~340 MB)
  char* ws = (char*)d_ws;
  size_t off = 0;
  auto carve = [&](size_t bytes) {
    void* p = ws + off;
    off += (bytes + 255) & ~(size_t)255;
    return p;
  };
  _Float16* WcpeH = (_Float16*)carve((size_t)KOFF * D * D * 2);
  _Float16* WlinH = (_Float16*)carve((size_t)D * D * 2);
  _Float16* WkH   = (_Float16*)carve((size_t)D * D * 2);
  _Float16* WvH   = (_Float16*)carve((size_t)D * D * 2);
  _Float16* xh    = (_Float16*)carve((size_t)T * D * 2);
  _Float16* kkh   = (_Float16*)carve((size_t)T * D * 2);
  _Float16* vvh   = (_Float16*)carve((size_t)T * D * 2);
  float*    qin   = (float*)carve((size_t)BSZ * VIEWS * D * 4);
  float*    qv    = (float*)carve((size_t)BSZ * VIEWS * D * 4);
  float*    sg    = (float*)carve((size_t)BSZ * HEADS * VIEWS * npts * 4);
  float*    oat   = (float*)carve((size_t)BSZ * VIEWS * D * 4);
  (void)ws_size; (void)n_in; (void)out_size;

  // weight conversion into WMMA-friendly tiled f16 layouts
  {
    int n1 = KOFF * D * D;
    k_conv_wcpe<<<(n1 + 255) / 256, 256, 0, stream>>>(Wcpe, WcpeH);
    int n2 = D * D;
    k_conv_tiled<<<(n2 + 255) / 256, 256, 0, stream>>>(Wlin, WlinH);
    k_conv_tiled<<<(n2 + 255) / 256, 256, 0, stream>>>(Wk, WkH);
    k_conv_tiled<<<(n2 + 255) / 256, 256, 0, stream>>>(Wv, WvH);
  }
  // fused cpe + linear + LN + residual
  k_cpe_ln<<<(T + ROWS - 1) / ROWS, 256, 0, stream>>>(feats, nbr_idx, nbr_valid, WcpeH, b_cpe,
                                                      WlinH, b_lin, ln_g, ln_b, xh, T);
  // K/V projections
  k_kv<<<(T + ROWS - 1) / ROWS, 256, 0, stream>>>(xh, WkH, bk, WvH, bv, kkh, vvh, T);
  // per-view pooled queries
  k_pool<<<BSZ * VIEWS, 256, 0, stream>>>(xh, vis, qin, npts);
  k_qproj<<<BSZ * VIEWS, 256, 0, stream>>>(qin, Wq, bq, qv);
  // attention
  k_attn<<<BSZ * HEADS * VIEWS, 256, 0, stream>>>(qv, kkh, vvh, vis, sg, oat, npts);
  // output projection
  k_cproj<<<BSZ * VIEWS, 256, 0, stream>>>(oat, Wc, bc, (float*)d_out);
}
